// CABgnn_39324720562991
// MI455X (gfx1250) — matrix operand order repaired and checked by reading, not compiled, not run on hardware
//
#include <hip/hip_runtime.h>
#include <hip/hip_bf16.h>

#define DIM 128
#define HID 256
// padded LDS row strides (elements) to avoid 16-way bank conflicts
#define AS  (DIM + 8)
#define HS  (HID + 8)
#define W1S (DIM + 8)   // sW1t rows are length DIM (k), one row per n
#define W2S (HID + 8)   // sW2t rows are length HID (k), one row per n

typedef __attribute__((ext_vector_type(16))) __bf16 bf16x16;
typedef __attribute__((ext_vector_type(8)))  float  f32x8;

union Frag {
    bf16x16        v;
    unsigned short s[16];
    unsigned int   u[8];
};

__device__ __forceinline__ unsigned short f2bf(float x) {
    union { float f; unsigned u; } v; v.f = x;
    unsigned r = v.u + 0x7FFFu + ((v.u >> 16) & 1u);   // round-to-nearest-even
    return (unsigned short)(r >> 16);
}

__device__ __forceinline__ float4 ld4(const float* p) { return *(const float4*)p; }
__device__ __forceinline__ void   st4(float* p, float4 v) { *(float4*)p = v; }
__device__ __forceinline__ float4 f4add(float4 a, float4 b) {
    return make_float4(a.x + b.x, a.y + b.y, a.z + b.z, a.w + b.w);
}

// ---------------------------------------------------------------------------
// x[i] = emb1[a0]+emb2[a1] for real nodes, teb for virtual nodes
// ---------------------------------------------------------------------------
__global__ void build_x_kernel(const int* __restrict__ xat,
                               const float* __restrict__ emb1,
                               const float* __restrict__ emb2,
                               const float* __restrict__ teb,
                               float* __restrict__ x, int N, int n) {
    int idx = blockIdx.x * 256 + threadIdx.x;           // over N*32 quads
    if (idx >= N * 32) return;
    int node = idx >> 5, f = (idx & 31) * 4;
    float4 v;
    if (node < n) {
        int a0 = xat[2 * node], a1 = xat[2 * node + 1];
        v = f4add(ld4(emb1 + (size_t)a0 * DIM + f), ld4(emb2 + (size_t)a1 * DIM + f));
    } else {
        v = ld4(teb + (size_t)(node - n) * DIM + f);
    }
    st4(x + (size_t)node * DIM + f, v);
}

// ---------------------------------------------------------------------------
// agg = x + (e1[4]+e2[0])
// ---------------------------------------------------------------------------
__global__ void agg_init_kernel(const float* __restrict__ x,
                                const float* __restrict__ e1L,
                                const float* __restrict__ e2L,
                                float* __restrict__ agg, int N) {
    int idx = blockIdx.x * 256 + threadIdx.x;
    if (idx >= N * 32) return;
    int node = idx >> 5, f = (idx & 31) * 4;
    float4 c = f4add(ld4(e1L + 4 * DIM + f), ld4(e2L + f));
    st4(agg + (size_t)node * DIM + f, f4add(ld4(x + (size_t)node * DIM + f), c));
}

// ---------------------------------------------------------------------------
// virtual-node edges: agg[i] += x[n+b]+vemb (plain); agg[n+b] += x[i]+vemb (atomic)
// vemb = e1[5] + e2[0]
// ---------------------------------------------------------------------------
__global__ void agg_vedges_kernel(const float* __restrict__ x,
                                  const int* __restrict__ batch,
                                  const float* __restrict__ e1L,
                                  const float* __restrict__ e2L,
                                  float* __restrict__ agg, int n) {
    int idx = blockIdx.x * 256 + threadIdx.x;           // over n*32 quads
    if (idx >= n * 32) return;
    int node = idx >> 5, f = (idx & 31) * 4;
    int vn = n + batch[node];
    float4 vemb = f4add(ld4(e1L + 5 * DIM + f), ld4(e2L + f));
    float4 xv = ld4(x + (size_t)vn * DIM + f);
    float4 xn = ld4(x + (size_t)node * DIM + f);
    // virtual -> node (exclusive per (node,f) thread, no atomics needed)
    float* pn = agg + (size_t)node * DIM + f;
    st4(pn, f4add(ld4(pn), f4add(xv, vemb)));
    // node -> virtual (many nodes per virtual node: atomic, lands in L2)
    float* pv = agg + (size_t)vn * DIM + f;
    float4 a = f4add(xn, vemb);
    atomicAdd(pv + 0, a.x); atomicAdd(pv + 1, a.y);
    atomicAdd(pv + 2, a.z); atomicAdd(pv + 3, a.w);
}

// ---------------------------------------------------------------------------
// real edges: agg[dst] += x[src] + e1[a0] + e2[a1]   (L2-resident atomics)
// ---------------------------------------------------------------------------
__global__ void agg_edges_kernel(const float* __restrict__ x,
                                 const int* __restrict__ ei,
                                 const int* __restrict__ ea,
                                 const float* __restrict__ e1L,
                                 const float* __restrict__ e2L,
                                 float* __restrict__ agg, int e) {
    int idx = blockIdx.x * 256 + threadIdx.x;           // over e*32 quads
    if (idx >= e * 32) return;
    int edge = idx >> 5, f = (idx & 31) * 4;
    int src = ei[edge], dst = ei[e + edge];
    int a0 = ea[2 * edge], a1 = ea[2 * edge + 1];
    float4 v = f4add(ld4(x + (size_t)src * DIM + f),
                     f4add(ld4(e1L + (size_t)a0 * DIM + f),
                           ld4(e2L + (size_t)a1 * DIM + f)));
    float* p = agg + (size_t)dst * DIM + f;
    atomicAdd(p + 0, v.x); atomicAdd(p + 1, v.y);
    atomicAdd(p + 2, v.z); atomicAdd(p + 3, v.w);
}

// ---------------------------------------------------------------------------
// MLP: y = relu(agg @ W1 + b1) @ W2 + b2   via V_WMMA_F32_16X16X32_BF16
// block = 256 threads = 8 waves; 128 rows / block (16 rows per wave)
// LDS: bf16 W1^T, W2^T, A tile, H tile (padded rows), ~234 KB of 320 KB/WGP
// ---------------------------------------------------------------------------
__global__ void __launch_bounds__(256, 1)
mlp_wmma_kernel(const float* __restrict__ agg,
                const float* __restrict__ W1, const float* __restrict__ b1,
                const float* __restrict__ W2, const float* __restrict__ b2,
                float* y, int N) {
    extern __shared__ unsigned short smem[];
    unsigned short* sW1t = smem;                        // [HID][W1S]  (n-major)
    unsigned short* sW2t = sW1t + HID * W1S;            // [DIM][W2S]  (n-major)
    unsigned short* sA   = sW2t + DIM * W2S;            // [128][AS]
    unsigned short* sH   = sA + 128 * AS;               // [128][HS]

    const int t = threadIdx.x;
    const int row0 = blockIdx.x * 128;

    for (int i = t; i < DIM * HID; i += 256) {          // W1 is [k][n]
        int k = i / HID, nn = i % HID;
        sW1t[nn * W1S + k] = f2bf(W1[i]);
    }
    for (int i = t; i < HID * DIM; i += 256) {          // W2 is [k][n]
        int k = i / DIM, nn = i % DIM;
        sW2t[nn * W2S + k] = f2bf(W2[i]);
    }
    for (int i = t; i < 128 * DIM; i += 256) {
        int r = i >> 7, c = i & 127;
        int gr = row0 + r;
        sA[r * AS + c] = f2bf(gr < N ? agg[(size_t)gr * DIM + c] : 0.0f);
    }
    // pull the NEXT block's A tile toward L2/WGP$ while this block computes
    // (emits global_prefetch_b8; speculative, safe at tail)
    {
        int pr = row0 + 128 + (t >> 1);
        if (pr < N)
            __builtin_prefetch(agg + (size_t)pr * DIM + (t & 1) * 64, 0, 1);
    }
    __syncthreads();

    const int lane = t & 31;
    const int half = lane >> 4;                         // wave32: lanes 0-15 / 16-31
    const int lm   = lane & 15;
    const int mrow = (t >> 5) * 16;                     // wave's row strip
    const unsigned short* aRow = sA + (mrow + lm) * AS;
    const unsigned short* hRow = sH + (mrow + lm) * HS;

    // ---- GEMM1: H(16x256) = relu(A(16x128) @ W1 + b1), bf16 out to LDS ----
    for (int nt = 0; nt < HID / 16; ++nt) {
        const int ncol = nt * 16 + lm;
        f32x8 acc = {0.f, 0.f, 0.f, 0.f, 0.f, 0.f, 0.f, 0.f};
        for (int kt = 0; kt < DIM / 32; ++kt) {
            Frag a, b;
            const int k0 = kt * 32;
#pragma unroll
            for (int j = 0; j < 8; ++j) {
                // A lane layout (ISA 7.12.2, 16-bit 16x32): VGPR j -> K pairs
                int ka = k0 + ((j < 4) ? 2 * j : 16 + 2 * (j - 4)) + half * 8;
                a.u[j] = *(const unsigned int*)(aRow + ka);
                // B lane layout: same K order as A, col = lm
                int kb = k0 + ((j < 4) ? 2 * j : 16 + 2 * (j - 4)) + half * 8;
                b.u[j] = *(const unsigned int*)(sW1t + ncol * W1S + kb);
            }
            acc = __builtin_amdgcn_wmma_f32_16x16x32_bf16(
                false, a.v, false, b.v, (short)0, acc, false, false);
        }
        const float bias = b1[ncol];
#pragma unroll
        for (int j = 0; j < 8; ++j) {                   // D: row = j + 8*half, col = lm
            float h = __builtin_fmaxf(acc[j] + bias, 0.f);   // v_max_num_f32
            sH[(mrow + j + half * 8) * HS + ncol] = f2bf(h);
        }
    }
    // wave-private H strip: in-wave DS ordering is enough, no barrier needed

    // ---- GEMM2: Y(16x128) = H(16x256) @ W2 + b2 ----
    for (int nt = 0; nt < DIM / 16; ++nt) {
        const int ncol = nt * 16 + lm;
        f32x8 acc = {0.f, 0.f, 0.f, 0.f, 0.f, 0.f, 0.f, 0.f};
        for (int kt = 0; kt < HID / 32; ++kt) {
            Frag a, b;
            const int k0 = kt * 32;
#pragma unroll
            for (int j = 0; j < 8; ++j) {
                int ka = k0 + ((j < 4) ? 2 * j : 16 + 2 * (j - 4)) + half * 8;
                a.u[j] = *(const unsigned int*)(hRow + ka);
                int kb = k0 + ((j < 4) ? 2 * j : 16 + 2 * (j - 4)) + half * 8;
                b.u[j] = *(const unsigned int*)(sW2t + ncol * W2S + kb);
            }
            acc = __builtin_amdgcn_wmma_f32_16x16x32_bf16(
                false, a.v, false, b.v, (short)0, acc, false, false);
        }
        const float bias = b2[ncol];
#pragma unroll
        for (int j = 0; j < 8; ++j) {
            int r = row0 + mrow + j + half * 8;
            if (r < N) y[(size_t)r * DIM + ncol] = acc[j] + bias;
        }
    }
}

// ---------------------------------------------------------------------------
// FiLM gamma/beta = task_embs @ W + b  (tiny GEMM, plain dot)
// ---------------------------------------------------------------------------
__global__ void film_gb_kernel(const float* __restrict__ te,
                               const float* __restrict__ Wg, const float* __restrict__ bg,
                               const float* __restrict__ Wb, const float* __restrict__ bb,
                               float* __restrict__ gamma, float* __restrict__ beta, int g) {
    int idx = blockIdx.x * 256 + threadIdx.x;
    if (idx >= g * DIM) return;
    int b = idx >> 7, f = idx & 127;
    const float* trow = te + (size_t)b * DIM;
    float sg = bg[f], sb = bb[f];
    for (int k = 0; k < DIM; ++k) {
        float tv = trow[k];
        sg += tv * Wg[k * DIM + f];
        sb += tv * Wb[k * DIM + f];
    }
    gamma[idx] = sg; beta[idx] = sb;
}

__global__ void film_apply_kernel(float* __restrict__ y,
                                  const float* __restrict__ gamma,
                                  const float* __restrict__ beta,
                                  const int* __restrict__ batch, int n) {
    int idx = blockIdx.x * 256 + threadIdx.x;           // over n*32 quads
    if (idx >= n * 32) return;
    int node = idx >> 5, f = (idx & 31) * 4;
    int b = batch[node];
    float4 ga = ld4(gamma + (size_t)b * DIM + f);
    float4 be = ld4(beta + (size_t)b * DIM + f);
    float* p = y + (size_t)node * DIM + f;
    float4 v = ld4(p);
    v.x = ga.x * v.x + be.x; v.y = ga.y * v.y + be.y;
    v.z = ga.z * v.z + be.z; v.w = ga.w * v.w + be.w;
    st4(p, v);
}

// ---------------------------------------------------------------------------
// batchnorm: per-feature sum / sumsq reduction, then scale/shift, then apply
// ---------------------------------------------------------------------------
__global__ void zero_stats_kernel(float* st) {
    if (threadIdx.x < 512) st[threadIdx.x] = 0.f;
}

__global__ void bn_stats_kernel(const float* __restrict__ y,
                                float* __restrict__ sum, float* __restrict__ sumsq, int N) {
    __shared__ float ls[256], lq[256];
    const int t = threadIdx.x;
    const int f = t & 127, h = t >> 7;
    float s = 0.f, q = 0.f;
    int r0 = blockIdx.x * 512;
    for (int r = r0 + h; r < r0 + 512 && r < N; r += 2) {
        float v = y[(size_t)r * DIM + f];
        s += v; q += v * v;
    }
    ls[t] = s; lq[t] = q;
    __syncthreads();
    if (h == 0) {
        atomicAdd(&sum[f], ls[t] + ls[t + 128]);
        atomicAdd(&sumsq[f], lq[t] + lq[t + 128]);
    }
}

__global__ void bn_final_kernel(const float* __restrict__ sum, const float* __restrict__ sumsq,
                                const float* __restrict__ w, const float* __restrict__ b,
                                float* __restrict__ scale, float* __restrict__ shift, int N) {
    int f = threadIdx.x;
    if (f >= 128) return;
    float inv = 1.0f / (float)N;
    float mu  = sum[f] * inv;
    float var = sumsq[f] * inv - mu * mu;
    float a   = w[f] * rsqrtf(var + 1e-5f);
    scale[f]  = a;
    shift[f]  = b[f] - mu * a;
}

__global__ void bn_apply_kernel(const float* __restrict__ y,
                                const float* __restrict__ scale,
                                const float* __restrict__ shift,
                                float* __restrict__ xout, float* __restrict__ outp,
                                int N, int n, int do_relu) {
    int idx = blockIdx.x * 256 + threadIdx.x;           // over N*32 quads
    if (idx >= N * 32) return;
    int node = idx >> 5, f = (idx & 31) * 4;
    float4 v = ld4(y + (size_t)node * DIM + f);
    float4 sc = ld4(scale + f), sh = ld4(shift + f);
    v.x = v.x * sc.x + sh.x; v.y = v.y * sc.y + sh.y;
    v.z = v.z * sc.z + sh.z; v.w = v.w * sc.w + sh.w;
    if (outp && node < n) st4(outp + (size_t)node * DIM + f, v);   // final layer output (no relu)
    if (do_relu) {
        v.x = __builtin_fmaxf(v.x, 0.f); v.y = __builtin_fmaxf(v.y, 0.f);
        v.z = __builtin_fmaxf(v.z, 0.f); v.w = __builtin_fmaxf(v.w, 0.f);
    }
    st4(xout + (size_t)node * DIM + f, v);
}

// ---------------------------------------------------------------------------
static inline int cdiv(long a, long b) { return (int)((a + b - 1) / b); }

extern "C" void kernel_launch(void* const* d_in, const int* in_sizes, int n_in,
                              void* d_out, int out_size, void* d_ws, size_t ws_size,
                              hipStream_t stream) {
    (void)n_in; (void)out_size; (void)ws_size;
    const int*   x_atoms  = (const int*)d_in[0];
    const int*   edge_idx = (const int*)d_in[1];
    const int*   edge_att = (const int*)d_in[2];
    const int*   batch    = (const int*)d_in[3];
    const float* task_e   = (const float*)d_in[6];
    const float* teb      = (const float*)d_in[7];
    const float* emb1     = (const float*)d_in[8];
    const float* emb2     = (const float*)d_in[9];
    const float* edge_e1  = (const float*)d_in[10];
    const float* edge_e2  = (const float*)d_in[11];
    const float* W1       = (const float*)d_in[12];
    const float* b1       = (const float*)d_in[13];
    const float* W2       = (const float*)d_in[14];
    const float* b2       = (const float*)d_in[15];
    const float* bn_w     = (const float*)d_in[16];
    const float* bn_b     = (const float*)d_in[17];
    const float* film_Wg  = (const float*)d_in[18];
    const float* film_bg  = (const float*)d_in[19];
    const float* film_Wb  = (const float*)d_in[20];
    const float* film_bb  = (const float*)d_in[21];

    const int n = in_sizes[0] / 2;          // 131072
    const int e = in_sizes[2] / 2;          // 524288
    const int g = in_sizes[6] / DIM;        // 4096
    const int N = n + g;                    // 135168

    // workspace carve (floats): X | AGG (aliased with MLP output) | gamma/beta | stats
    float* X   = (float*)d_ws;
    float* AGG = X + (size_t)N * DIM;
    float* GB  = AGG + (size_t)N * DIM;     // [fl][{gamma,beta}][g][DIM]
    float* ST  = GB + (size_t)4 * g * DIM;  // sum|sumsq|scale|shift = 512 floats

    const int T = 256;
    const size_t smem = (size_t)(HID * W1S + DIM * W2S + 128 * AS + 128 * HS) * 2;
    hipFuncSetAttribute((const void*)mlp_wmma_kernel,
                        hipFuncAttributeMaxDynamicSharedMemorySize, (int)smem);

    // node features
    build_x_kernel<<<cdiv((long)N * 32, T), T, 0, stream>>>(x_atoms, emb1, emb2, teb, X, N, n);

    // FiLM gamma/beta for both film stages (fl = 0 -> layer 1, fl = 1 -> layer 3)
    for (int fl = 0; fl < 2; ++fl) {
        film_gb_kernel<<<cdiv((long)g * DIM, T), T, 0, stream>>>(
            task_e, film_Wg + (size_t)fl * DIM * DIM, film_bg + (size_t)fl * DIM,
            film_Wb + (size_t)fl * DIM * DIM, film_bb + (size_t)fl * DIM,
            GB + (size_t)fl * 2 * g * DIM, GB + (size_t)fl * 2 * g * DIM + (size_t)g * DIM, g);
    }

    for (int L = 0; L < 4; ++L) {
        const float* e1L = edge_e1 + (size_t)L * 6 * DIM;
        const float* e2L = edge_e2 + (size_t)L * 3 * DIM;

        // aggregation: agg = x + c + segment_sum(x[src]+emb, dst)
        agg_init_kernel<<<cdiv((long)N * 32, T), T, 0, stream>>>(X, e1L, e2L, AGG, N);
        agg_vedges_kernel<<<cdiv((long)n * 32, T), T, 0, stream>>>(X, batch, e1L, e2L, AGG, n);
        agg_edges_kernel<<<cdiv((long)e * 32, T), T, 0, stream>>>(X, edge_idx, edge_att,
                                                                  e1L, e2L, AGG, e);
        // MLP via bf16 WMMA (in-place: reads staged to LDS before block writes)
        mlp_wmma_kernel<<<cdiv(N, 128), T, smem, stream>>>(
            AGG, W1 + (size_t)L * DIM * HID, b1 + (size_t)L * HID,
            W2 + (size_t)L * HID * DIM, b2 + (size_t)L * DIM, AGG, N);

        if (L == 1 || L == 3) {
            int fl = L / 3;
            film_apply_kernel<<<cdiv((long)n * 32, T), T, 0, stream>>>(
                AGG, GB + (size_t)fl * 2 * g * DIM,
                GB + (size_t)fl * 2 * g * DIM + (size_t)g * DIM, batch, n);
        }

        // batchnorm + (relu except last layer); last layer streams rows [0,n) to d_out
        zero_stats_kernel<<<1, 512, 0, stream>>>(ST);
        bn_stats_kernel<<<cdiv(N, 512), T, 0, stream>>>(AGG, ST, ST + 128, N);
        bn_final_kernel<<<1, 128, 0, stream>>>(ST, ST + 128, bn_w + (size_t)L * DIM,
                                               bn_b + (size_t)L * DIM, ST + 256, ST + 384, N);
        bn_apply_kernel<<<cdiv((long)N * 32, T), T, 0, stream>>>(
            AGG, ST + 256, ST + 384, X, (L == 3) ? (float*)d_out : nullptr, N, n, L != 3);
    }
}
